// RWKV_6_50448685859314
// MI455X (gfx1250) — compile-verified, matrix operand order checked by reading, not run on hardware
//
#include <hip/hip_runtime.h>
#include <hip/hip_bf16.h>

typedef __attribute__((ext_vector_type(16))) __bf16 v16bf;
typedef __attribute__((ext_vector_type(8)))  float  v8f;
typedef __attribute__((ext_vector_type(2)))  float  v2f;

#define B_   4
#define T_   2048
#define C_   2048
#define H_   32
#define K_   64
#define NTOK (B_ * T_)   // 8192

// ---------------------------------------------------------------------------
// Elementwise converts
// ---------------------------------------------------------------------------
__global__ void cvt_bf16_kernel(const float* __restrict__ src, __bf16* __restrict__ dst, int n) {
    int i = blockIdx.x * 256 + threadIdx.x;
    if (i < n) dst[i] = (__bf16)src[i];
}

// dst[c][r] = src[r][c]  (R x Cc -> Cc x R), bf16
__global__ void cvtT_bf16_kernel(const float* __restrict__ src, __bf16* __restrict__ dst, int R, int Cc) {
    int i = blockIdx.x * 256 + threadIdx.x;
    if (i < R * Cc) {
        int r = i / Cc, c = i % Cc;
        dst[(size_t)c * R + r] = (__bf16)src[i];
    }
}

// ---------------------------------------------------------------------------
// LayerNorm + token shift + xxx
// ---------------------------------------------------------------------------
__device__ inline float block_reduce_sum(float v, float* sbuf) {
    int tid = threadIdx.x;
    sbuf[tid] = v;
    __syncthreads();
    for (int s = 128; s > 0; s >>= 1) {
        if (tid < s) sbuf[tid] += sbuf[tid + s];
        __syncthreads();
    }
    float r = sbuf[0];
    __syncthreads();
    return r;
}

__global__ __launch_bounds__(256) void ln_shift_kernel(
    const float* __restrict__ x, const float* __restrict__ state_shift,
    const float* __restrict__ lnw, const float* __restrict__ lnb,
    const float* __restrict__ maa_x,
    float* __restrict__ xn_out, float* __restrict__ dx_out, __bf16* __restrict__ xxx_out)
{
    __shared__ float sbuf[256];
    int n = blockIdx.x;
    int b = n / T_, t = n % T_;
    const float* xr = x + (size_t)n * C_;
    const float* xp = xr - C_;   // row t-1 (only dereferenced when t > 0)
    int tid = threadIdx.x;

    float s1 = 0.f, s2 = 0.f, s3 = 0.f, s4 = 0.f;
    for (int c = tid; c < C_; c += 256) {
        float a = xr[c]; s1 += a; s2 += a * a;
        if (t > 0) { float p = xp[c]; s3 += p; s4 += p * p; }
    }
    float m   = block_reduce_sum(s1, sbuf) * (1.f / C_);
    float ms  = block_reduce_sum(s2, sbuf) * (1.f / C_);
    float inv = rsqrtf(ms - m * m + 1e-5f);
    float mp = 0.f, invp = 0.f;
    if (t > 0) {  // uniform branch per block
        mp = block_reduce_sum(s3, sbuf) * (1.f / C_);
        float msp = block_reduce_sum(s4, sbuf) * (1.f / C_);
        invp = rsqrtf(msp - mp * mp + 1e-5f);
    }
    for (int c = tid; c < C_; c += 256) {
        float w = lnw[c], bb = lnb[c];
        float xnv = (xr[c] - m) * inv * w + bb;
        float sh  = (t > 0) ? ((xp[c] - mp) * invp * w + bb)
                            : state_shift[(size_t)b * C_ + c];
        float dxv = sh - xnv;
        size_t idx = (size_t)n * C_ + c;
        xn_out[idx]  = xnv;
        dx_out[idx]  = dxv;
        xxx_out[idx] = (__bf16)(xnv + dxv * maa_x[c]);
    }
}

// ---------------------------------------------------------------------------
// WMMA bf16 GEMM: out[M,N] = A[M,K] @ Bw[N,K]^T   (both K-contiguous "TN")
// Double-buffered LDS tiles filled with async-to-LDS copies (ASYNCcnt).
// ---------------------------------------------------------------------------
#define BM  128
#define BN  128
#define BK  32
#define LDT 40     // LDS row stride (bf16 elems): 80B, 16B-aligned, conflict-free

#define EP_F32       0
#define EP_TANH_BF16 1
#define EP_EW        2
#define EP_SILU      3

// one 16-byte lane-wise async copy: LDS[lds_addr] = MEM[gptr]
#define ASYNC_CP16(lds_addr, gptr)                                              \
    asm volatile("global_load_async_to_lds_b128 %0, %1, off"                    \
                 :: "v"(lds_addr), "v"(gptr) : "memory")

__global__ __launch_bounds__(256) void gemm_bf16_kernel(
    const __bf16* __restrict__ A, const __bf16* __restrict__ Bw,
    void* __restrict__ outp, const float* __restrict__ aux,
    int M, int N, int Kd, int mode)
{
    __shared__ __bf16 As[2][BM * LDT];
    __shared__ __bf16 Bs[2][BN * LDT];

    int tid  = threadIdx.x;
    int lane = tid & 31;
    int wave = tid >> 5;      // 0..7
    int wm   = wave & 3;      // wave row 0..3  (32 rows each)
    int wn   = wave >> 2;     // wave col 0..1  (64 cols each)
    int bm   = blockIdx.x * BM;
    int bn   = blockIdx.y * BN;

    v8f zero = {0.f, 0.f, 0.f, 0.f, 0.f, 0.f, 0.f, 0.f};
    v8f acc[2][4];
    for (int i = 0; i < 2; i++)
        for (int j = 0; j < 4; j++) acc[i][j] = zero;

    int lrow = tid >> 1;            // 0..127
    int lcol = (tid & 1) * 16;      // 0 or 16

    // per-lane source rows; clamp B row so every wave issues the same number
    // of async ops (uniform ASYNCcnt), stores are guarded instead
    const __bf16* aSrc = A  + (size_t)(bm + lrow) * Kd + lcol;
    int nrowc = bn + lrow; if (nrowc > N - 1) nrowc = N - 1;
    const __bf16* bSrc = Bw + (size_t)nrowc * Kd + lcol;

    unsigned aDst[2], bDst[2];
    #pragma unroll
    for (int p = 0; p < 2; p++) {
        aDst[p] = (unsigned)(uintptr_t)&As[p][lrow * LDT + lcol];
        bDst[p] = (unsigned)(uintptr_t)&Bs[p][lrow * LDT + lcol];
    }

    const int half = lane >> 4;     // 16-bit A/B fragment layout (ISA 7.12.2)
    const int fr   = lane & 15;
    const int k0   = half * 8;      // lanes 0-15: K 0..7/16..23 ; lanes 16-31: K 8..15/24..31

    union Frag { uint4 q[2]; v16bf v; };

    const int nt = Kd / BK;
    // prologue: fill buffer 0 with tile 0 (4 async ops per wave)
    ASYNC_CP16(aDst[0],      aSrc);
    ASYNC_CP16(aDst[0] + 16, aSrc + 8);
    ASYNC_CP16(bDst[0],      bSrc);
    ASYNC_CP16(bDst[0] + 16, bSrc + 8);

    for (int it = 0; it < nt; it++) {
        int p = it & 1;
        if (it + 1 < nt) {          // issue tile it+1 into the other buffer
            int kt = (it + 1) * BK;
            ASYNC_CP16(aDst[p ^ 1],      aSrc + kt);
            ASYNC_CP16(aDst[p ^ 1] + 16, aSrc + kt + 8);
            ASYNC_CP16(bDst[p ^ 1],      bSrc + kt);
            ASYNC_CP16(bDst[p ^ 1] + 16, bSrc + kt + 8);
            asm volatile("s_wait_asynccnt 0x4" ::: "memory");  // tile it resident
        } else {
            asm volatile("s_wait_asynccnt 0x0" ::: "memory");
        }
        __syncthreads();            // every wave's tile-it copies are in LDS

        Frag a[2], b[4];
        #pragma unroll
        for (int i = 0; i < 2; i++) {
            const __bf16* pp = &As[p][(wm * 32 + i * 16 + fr) * LDT + k0];
            a[i].q[0] = *(const uint4*)pp;
            a[i].q[1] = *(const uint4*)(pp + 16);
        }
        #pragma unroll
        for (int j = 0; j < 4; j++) {
            const __bf16* pp = &Bs[p][(wn * 64 + j * 16 + fr) * LDT + k0];
            b[j].q[0] = *(const uint4*)pp;
            b[j].q[1] = *(const uint4*)(pp + 16);
        }
        #pragma unroll
        for (int i = 0; i < 2; i++)
            #pragma unroll
            for (int j = 0; j < 4; j++)
                acc[i][j] = __builtin_amdgcn_wmma_f32_16x16x32_bf16(
                    false, a[i].v, false, b[j].v, (short)0, acc[i][j], false, false);
        __syncthreads();            // done reading buf p; next issue may overwrite it
    }

    // Epilogue: C layout — lane<16: col=lane, rows r; lane>=16: col=lane-16, rows r+8
    #pragma unroll
    for (int i = 0; i < 2; i++) {
        #pragma unroll
        for (int j = 0; j < 4; j++) {
            int row0 = bm + wm * 32 + i * 16 + (half ? 8 : 0);
            int col  = bn + wn * 64 + j * 16 + fr;
            if (col < N) {
                #pragma unroll
                for (int r = 0; r < 8; r++) {
                    float val = acc[i][j][r];
                    size_t idx = (size_t)(row0 + r) * N + col;
                    if (mode == EP_TANH_BF16)      ((__bf16*)outp)[idx] = (__bf16)tanhf(val);
                    else if (mode == EP_EW)        ((float*)outp)[idx]  = expf(-expf(aux[col] + val));
                    else if (mode == EP_SILU)      ((float*)outp)[idx]  = val / (1.f + expf(-val));
                    else                           ((float*)outp)[idx]  = val;
                }
            }
        }
    }
}

// ---------------------------------------------------------------------------
// mix @ maa_w2 (K=32, tiny) + combine: x{w,k,v,r,g} = (mix2+wkvrg)*dx + xn
// ---------------------------------------------------------------------------
__global__ __launch_bounds__(256) void mix_combine_kernel(
    const __bf16* __restrict__ mix, const float* __restrict__ w2,
    const float* __restrict__ wkvrg,
    const float* __restrict__ xn, const float* __restrict__ dx,
    __bf16* __restrict__ xw, __bf16* __restrict__ xk, __bf16* __restrict__ xv,
    __bf16* __restrict__ xr, __bf16* __restrict__ xg)
{
    __shared__ float mrow[160];
    int n = blockIdx.x, tid = threadIdx.x;
    if (tid < 160) mrow[tid] = (float)mix[(size_t)n * 160 + tid];
    __syncthreads();
    __bf16* outs[5] = {xw, xk, xv, xr, xg};
    for (int c = tid; c < C_; c += 256) {
        size_t idx = (size_t)n * C_ + c;
        float dxv = dx[idx], xnv = xn[idx];
        #pragma unroll
        for (int f = 0; f < 5; f++) {
            float acc = 0.f;
            #pragma unroll
            for (int d = 0; d < 32; d++)
                acc += mrow[f * 32 + d] * w2[((size_t)(f * 32 + d)) * C_ + c];
            outs[f][idx] = (__bf16)((acc + wkvrg[f * C_ + c]) * dxv + xnv);
        }
    }
}

// ---------------------------------------------------------------------------
// Chunked WKV scan with f32 WMMA (V_WMMA_F32_16X16X4_F32).
// One block (256 thr = 8 waves) per (b,h). Chunk L=16, head dim 64.
//   P_t[k]   = prod_{s<t} ew_s[k]        (P_0 = 1)
//   q~_t     = q_t * P_t ;  k~_s = k_s / P_{s+1}
//   S        = Q~ K~^T, strictly-lower mask, diag_t = sum_k q_t u_k k_t
//   O        = Q~ @ H0 + S' @ V
//   H        = P_L .* (H0 + K~^T @ V)
// Waves 0-3: O columns [16w,16w+16).  Wave 4: S (+mask+diag).  Waves 4-7: K~^T V.
// ---------------------------------------------------------------------------
#define CL 16
__global__ __launch_bounds__(256) void wkv_chunk_kernel(
    const float* __restrict__ rq, const float* __restrict__ kk_,
    const float* __restrict__ vv_, const float* __restrict__ ew,
    const float* __restrict__ g, const float* __restrict__ u,
    const float* __restrict__ h0, float* __restrict__ out, float* __restrict__ hout)
{
    __shared__ float sq [CL][64], sk0[CL][64], sv[CL][64], sw[CL][64];
    __shared__ float sqt[CL][64], skt[CL][64];                 // q~, k~
    __shared__ float sh [64][64];                              // running state H
    __shared__ float ss [CL][18];                              // masked S' (stride 18: 8B-aligned rows)
    __shared__ float spl[64], sdiag[CL], su[64];

    int bh = blockIdx.x;
    int b = bh / H_, h = bh % H_;
    int tid  = threadIdx.x;
    int lane = tid & 31;
    int wave = tid >> 5;       // 0..7
    const int fr = lane & 15;  // M/N index inside a 16x16 tile
    const int hf = lane >> 4;  // selects K pair {2hf, 2hf+1} within a K=4 step

    // init state + u
    const float* h0p = h0 + (size_t)bh * K_ * K_;
    for (int i = tid; i < 64 * 64; i += 256) sh[i >> 6][i & 63] = h0p[i];
    if (tid < 64) su[tid] = u[h * K_ + tid];

    size_t base = (size_t)b * T_ * C_ + h * K_;
    v8f zero = {0.f, 0.f, 0.f, 0.f, 0.f, 0.f, 0.f, 0.f};

    for (int t0 = 0; t0 < T_; t0 += CL) {
        __syncthreads();   // prev chunk fully consumed (incl. sh merge, sv reads)

        // ---- phase 0: stage chunk tiles
        for (int i = tid; i < CL * 64; i += 256) {
            int s = i >> 6, c = i & 63;
            size_t gi = base + (size_t)(t0 + s) * C_ + c;
            sq [s][c] = rq[gi];
            sk0[s][c] = kk_[gi];
            sv [s][c] = vv_[gi];
            sw [s][c] = ew[gi];
        }
        __syncthreads();

        // ---- phase 1: cumulative decay, scaled q~/k~, P_L, diag
        if (tid < 64) {
            int k = tid;
            float P = 1.f;
            #pragma unroll
            for (int t = 0; t < CL; t++) {
                sqt[t][k] = sq[t][k] * P;    // q~_t = q_t * P_t
                P *= sw[t][k];               // P_{t+1}
                skt[t][k] = sk0[t][k] / P;   // k~_t = k_t / P_{t+1}
            }
            spl[k] = P;                      // P_L
        } else if (tid < 80) {
            int t = tid - 64;
            float D = 0.f;
            for (int k = 0; k < 64; k++) D += sq[t][k] * su[k] * sk0[t][k];
            sdiag[t] = D;
        }
        __syncthreads();

        // ---- phase 2a
        v8f oacc = zero;           // waves 0-3: O tile accum
        v8f hacc[4];               // waves 4-7: K~^T V tiles (mi = wave-4, ni = 0..3)
        #pragma unroll
        for (int j = 0; j < 4; j++) hacc[j] = zero;

        if (wave < 4) {
            // O1 = Q~ @ H0, columns [16*wave, +16)
            int n0 = wave * 16;
            #pragma unroll
            for (int kt = 0; kt < 16; kt++) {
                int ak = kt * 4 + 2 * hf;
                v2f a, bvec;
                float2 af = *(const float2*)&sqt[fr][ak];
                a[0] = af.x; a[1] = af.y;
                bvec[0] = sh[ak][n0 + fr];
                bvec[1] = sh[ak + 1][n0 + fr];
                oacc = __builtin_amdgcn_wmma_f32_16x16x4_f32(
                    false, a, false, bvec, (short)0, oacc, false, false);
            }
        } else {
            if (wave == 4) {
                // S = Q~ K~^T  (16x16, K=64)
                v8f sacc = zero;
                #pragma unroll
                for (int kt = 0; kt < 16; kt++) {
                    int ak = kt * 4 + 2 * hf;
                    v2f a, bvec;
                    float2 af = *(const float2*)&sqt[fr][ak];
                    a[0] = af.x; a[1] = af.y;
                    float2 bf = *(const float2*)&skt[fr][ak];
                    bvec[0] = bf.x; bvec[1] = bf.y;
                    sacc = __builtin_amdgcn_wmma_f32_16x16x4_f32(
                        false, a, false, bvec, (short)0, sacc, false, false);
                }
                // mask + diag, store S' (C layout: col s = fr, row t = r + 8*hf)
                #pragma unroll
                for (int r = 0; r < 8; r++) {
                    int t = r + 8 * hf, s = fr;
                    float val = sacc[r];
                    if (s > t) val = 0.f;
                    else if (s == t) val = sdiag[t];
                    ss[t][s] = val;
                }
            }
            // K~^T @ V : wave handles row-group mi = wave-4, all 4 col-groups
            int mi = wave - 4;
            #pragma unroll
            for (int ni = 0; ni < 4; ni++) {
                #pragma unroll
                for (int kt = 0; kt < 4; kt++) {
                    int ak = kt * 4 + 2 * hf;       // time index pair
                    v2f a, bvec;
                    a[0] = skt[ak][mi * 16 + fr];      // A[m][time] = k~_time[feat m]
                    a[1] = skt[ak + 1][mi * 16 + fr];
                    bvec[0] = sv[ak][ni * 16 + fr];    // B[time][v]
                    bvec[1] = sv[ak + 1][ni * 16 + fr];
                    hacc[ni] = __builtin_amdgcn_wmma_f32_16x16x4_f32(
                        false, a, false, bvec, (short)0, hacc[ni], false, false);
                }
            }
        }
        __syncthreads();   // S' stored; Q~@H0 done reading sh; hacc in regs

        // ---- phase 2b
        if (wave < 4) {
            // O += S' @ V (K=16), then gate + store
            int n0 = wave * 16;
            #pragma unroll
            for (int kt = 0; kt < 4; kt++) {
                int ak = kt * 4 + 2 * hf;
                v2f a, bvec;
                float2 af = *(const float2*)&ss[fr][ak];
                a[0] = af.x; a[1] = af.y;
                bvec[0] = sv[ak][n0 + fr];
                bvec[1] = sv[ak + 1][n0 + fr];
                oacc = __builtin_amdgcn_wmma_f32_16x16x4_f32(
                    false, a, false, bvec, (short)0, oacc, false, false);
            }
            #pragma unroll
            for (int r = 0; r < 8; r++) {
                int t = r + 8 * hf;
                int col = n0 + fr;
                size_t gi = base + (size_t)(t0 + t) * C_ + col;
                out[gi] = oacc[r] * g[gi];
            }
        } else {
            // H0 = P_L .* (H0 + K~^T V)   (rows [16*mi, +16))
            int mi = wave - 4;
            #pragma unroll
            for (int ni = 0; ni < 4; ni++) {
                #pragma unroll
                for (int r = 0; r < 8; r++) {
                    int feat = mi * 16 + r + 8 * hf;
                    int vc   = ni * 16 + fr;
                    sh[feat][vc] = spl[feat] * (sh[feat][vc] + hacc[ni][r]);
                }
            }
        }
    }

    __syncthreads();
    float* hop = hout + (size_t)bh * K_ * K_;
    for (int i = tid; i < 64 * 64; i += 256) hop[i] = sh[i >> 6][i & 63];
}

// ---------------------------------------------------------------------------
// Host launch
// ---------------------------------------------------------------------------
extern "C" void kernel_launch(void* const* d_in, const int* in_sizes, int n_in,
                              void* d_out, int out_size, void* d_ws, size_t ws_size,
                              hipStream_t stream)
{
    const float* x           = (const float*)d_in[0];
    const float* state_shift = (const float*)d_in[1];
    const float* state_wkv   = (const float*)d_in[2];
    const float* maa_x       = (const float*)d_in[3];
    const float* wkvrg       = (const float*)d_in[4];
    const float* maa_w1      = (const float*)d_in[5];
    const float* maa_w2      = (const float*)d_in[6];
    const float* decay_w1    = (const float*)d_in[7];
    const float* decay_w2    = (const float*)d_in[8];
    const float* time_decay  = (const float*)d_in[9];
    const float* wR          = (const float*)d_in[10];
    const float* wK          = (const float*)d_in[11];
    const float* wV          = (const float*)d_in[12];
    const float* wG          = (const float*)d_in[13];
    const float* lnw         = (const float*)d_in[14];
    const float* lnb         = (const float*)d_in[15];
    const float* u           = (const float*)d_in[16];

    char* wsb = (char*)d_ws;
    size_t off = 0;
    auto alloc = [&](size_t bytes) -> void* {
        void* p = wsb + off;
        off = (off + bytes + 255) & ~(size_t)255;
        return p;
    };
    const size_t NC = (size_t)NTOK * C_;

    float*  xn   = (float*) alloc(NC * 4);
    float*  dxb  = (float*) alloc(NC * 4);
    __bf16* xxx  = (__bf16*)alloc(NC * 2);
    __bf16* mix  = (__bf16*)alloc((size_t)NTOK * 160 * 2);
    __bf16* xw   = (__bf16*)alloc(NC * 2);
    __bf16* xk   = (__bf16*)alloc(NC * 2);
    __bf16* xv   = (__bf16*)alloc(NC * 2);
    __bf16* xr   = (__bf16*)alloc(NC * 2);
    __bf16* xg   = (__bf16*)alloc(NC * 2);
    __bf16* t1   = (__bf16*)alloc((size_t)NTOK * 64 * 2);
    float*  ewb  = (float*) alloc(NC * 4);
    float*  rb   = (float*) alloc(NC * 4);
    float*  kb   = (float*) alloc(NC * 4);
    float*  vb   = (float*) alloc(NC * 4);
    float*  gb   = (float*) alloc(NC * 4);
    __bf16* wRb  = (__bf16*)alloc((size_t)C_ * C_ * 2);
    __bf16* wKb  = (__bf16*)alloc((size_t)C_ * C_ * 2);
    __bf16* wVb  = (__bf16*)alloc((size_t)C_ * C_ * 2);
    __bf16* wGb  = (__bf16*)alloc((size_t)C_ * C_ * 2);
    __bf16* w1t  = (__bf16*)alloc((size_t)160 * C_ * 2);   // maa_w1^T  (160 x 2048)
    __bf16* dw1t = (__bf16*)alloc((size_t)64 * C_ * 2);    // decay_w1^T (64 x 2048)
    __bf16* dw2t = (__bf16*)alloc((size_t)C_ * 64 * 2);    // decay_w2^T (2048 x 64)

    const int CC = C_ * C_;
    // 1) weight conversions
    cvt_bf16_kernel<<<(CC + 255) / 256, 256, 0, stream>>>(wR, wRb, CC);
    cvt_bf16_kernel<<<(CC + 255) / 256, 256, 0, stream>>>(wK, wKb, CC);
    cvt_bf16_kernel<<<(CC + 255) / 256, 256, 0, stream>>>(wV, wVb, CC);
    cvt_bf16_kernel<<<(CC + 255) / 256, 256, 0, stream>>>(wG, wGb, CC);
    cvtT_bf16_kernel<<<(C_ * 160 + 255) / 256, 256, 0, stream>>>(maa_w1, w1t, C_, 160);
    cvtT_bf16_kernel<<<(C_ * 64  + 255) / 256, 256, 0, stream>>>(decay_w1, dw1t, C_, 64);
    cvtT_bf16_kernel<<<(64 * C_  + 255) / 256, 256, 0, stream>>>(decay_w2, dw2t, 64, C_);

    // 2) LN + shift + xxx
    ln_shift_kernel<<<NTOK, 256, 0, stream>>>(x, state_shift, lnw, lnb, maa_x, xn, dxb, xxx);

    // 3) mix = tanh(xxx @ maa_w1)  : M=8192 N=160 K=2048
    gemm_bf16_kernel<<<dim3(NTOK / BM, (160 + BN - 1) / BN), 256, 0, stream>>>(
        xxx, w1t, mix, nullptr, NTOK, 160, C_, EP_TANH_BF16);

    // 4) mix2 + combine -> xw,xk,xv,xr,xg
    mix_combine_kernel<<<NTOK, 256, 0, stream>>>(mix, maa_w2, wkvrg, xn, dxb, xw, xk, xv, xr, xg);

    // 5) t1 = tanh(xw @ decay_w1)  : M=8192 N=64 K=2048
    gemm_bf16_kernel<<<dim3(NTOK / BM, 1), 256, 0, stream>>>(
        xw, dw1t, t1, nullptr, NTOK, 64, C_, EP_TANH_BF16);

    // 6) ew = exp(-exp(time_decay + t1 @ decay_w2)) : M=8192 N=2048 K=64
    gemm_bf16_kernel<<<dim3(NTOK / BM, C_ / BN), 256, 0, stream>>>(
        t1, dw2t, ewb, time_decay, NTOK, C_, 64, EP_EW);

    // 7) r, k, v, g projections : M=8192 N=2048 K=2048
    gemm_bf16_kernel<<<dim3(NTOK / BM, C_ / BN), 256, 0, stream>>>(
        xr, wRb, rb, nullptr, NTOK, C_, C_, EP_F32);
    gemm_bf16_kernel<<<dim3(NTOK / BM, C_ / BN), 256, 0, stream>>>(
        xk, wKb, kb, nullptr, NTOK, C_, C_, EP_F32);
    gemm_bf16_kernel<<<dim3(NTOK / BM, C_ / BN), 256, 0, stream>>>(
        xv, wVb, vb, nullptr, NTOK, C_, C_, EP_F32);
    gemm_bf16_kernel<<<dim3(NTOK / BM, C_ / BN), 256, 0, stream>>>(
        xg, wGb, gb, nullptr, NTOK, C_, C_, EP_SILU);

    // 8) chunked WKV scan (f32 WMMA) + gate, writes out and h_final
    float* outp = (float*)d_out;
    float* hout = outp + NC;
    wkv_chunk_kernel<<<B_ * H_, 256, 0, stream>>>(rb, kb, vb, ewb, gb, u, state_wkv, outp, hout);
}
